// QuantumKernelMethod_65481071407866
// MI455X (gfx1250) — matrix operand order repaired and checked by reading, not compiled, not run on hardware
//
#include <hip/hip_runtime.h>
#include <hip/hip_bf16.h>

// ---------------------------------------------------------------------------
// RBF kernel K[i,j] = exp(-gamma * ||x_i - y_j||^2), M=N=8192, D=64, fp32.
// MI455X strategy:
//   - store floor: 256MiB / 23.3TB/s ~ 11.5us -> WMMA for the 8.6 GFLOP GEMM.
//   - prep: fp32 -> fp16 hi + fp16 lo (22 mantissa bits) + exact fp32 norms,
//     staged once in d_ws (~4.3MB, L2/WGP$-resident).
//   - main: dot(x,y) via 3-term split on v_wmma_f32_16x16x32_f16
//     (xh*yh + xh*yl + xl*yh); epilogue folded to fma+min+v_exp_f32:
//     exp(-g*max(sq,0)) = exp2(min(2g*log2e*dot - g*log2e*(xsq+ysq), 0)).
//   - all Y/out addressing uses constant instruction offsets (24-bit field),
//     hoisting 64-bit address math out of the tile loop.
// ---------------------------------------------------------------------------

typedef _Float16 half4  __attribute__((ext_vector_type(4)));
typedef _Float16 half8  __attribute__((ext_vector_type(8)));
typedef _Float16 v16h   __attribute__((ext_vector_type(16)));
typedef float    v8f    __attribute__((ext_vector_type(8)));

#define MDIM 8192
#define NDIM 8192
#define DDIM 64

// -------------------- prep: fp32 -> (f16 hi, f16 lo) + row norms ------------
__global__ __launch_bounds__(256) void rbf_prep(
    const float* __restrict__ X, const float* __restrict__ Y,
    float* __restrict__ xsq, float* __restrict__ ysq,
    _Float16* __restrict__ Xh, _Float16* __restrict__ Xl,
    _Float16* __restrict__ Yh, _Float16* __restrict__ Yl)
{
    int r = blockIdx.x * 256 + threadIdx.x;
    const float* src;
    _Float16 *h, *l;
    float* nq;
    if (r < MDIM) {
        src = X + (size_t)r * DDIM;
        h = Xh + (size_t)r * DDIM; l = Xl + (size_t)r * DDIM;
        nq = xsq + r;
    } else if (r < MDIM + NDIM) {
        int rr = r - MDIM;
        src = Y + (size_t)rr * DDIM;
        h = Yh + (size_t)rr * DDIM; l = Yl + (size_t)rr * DDIM;
        nq = ysq + rr;
    } else {
        return;
    }
    float s = 0.0f;
#pragma unroll 4
    for (int i = 0; i < DDIM; i += 4) {
        float4 f = *(const float4*)(src + i);
        half4 h4, l4;
        h4.x = (_Float16)f.x; l4.x = (_Float16)(f.x - (float)h4.x);
        h4.y = (_Float16)f.y; l4.y = (_Float16)(f.y - (float)h4.y);
        h4.z = (_Float16)f.z; l4.z = (_Float16)(f.z - (float)h4.z);
        h4.w = (_Float16)f.w; l4.w = (_Float16)(f.w - (float)h4.w);
        *(half4*)(h + i) = h4;
        *(half4*)(l + i) = l4;
        s = fmaf(f.x, f.x, s);
        s = fmaf(f.y, f.y, s);
        s = fmaf(f.z, f.z, s);
        s = fmaf(f.w, f.w, s);
    }
    *nq = s;
}

// load 2x 8 contiguous halves and concatenate into a v16h WMMA operand
__device__ __forceinline__ v16h ld16(const _Float16* p, int off0, int off1) {
    half8 a = *(const half8*)(p + off0);
    half8 b = *(const half8*)(p + off1);
    return __builtin_shufflevector(a, b, 0,1,2,3,4,5,6,7,8,9,10,11,12,13,14,15);
}

// -------------------- main: WMMA GEMM + RBF epilogue ------------------------
// block = 256 threads = 8 waves; wave w computes output rows
// [blockIdx.y*128 + 16w, +16) x cols [blockIdx.x*64, +64) as 4 WMMA tiles.
__global__ __launch_bounds__(256) void rbf_main(
    const _Float16* __restrict__ Xh, const _Float16* __restrict__ Xl,
    const _Float16* __restrict__ Yh, const _Float16* __restrict__ Yl,
    const float* __restrict__ xsq, const float* __restrict__ ysq,
    const float* __restrict__ gamma, float* __restrict__ out)
{
    const int lane  = threadIdx.x & 31;
    const int wave  = threadIdx.x >> 5;
    const int l15   = lane & 15;
    const int hi16  = (lane >> 4) & 1;

    const int mBase = blockIdx.y * 128 + wave * 16;
    const int nBase = blockIdx.x * 64;

    const float g   = gamma[0];
    const float l2e = 1.4426950408889634f;   // log2(e)
    const float gl  = -g * l2e;              // -g*log2e
    const float k2  = 2.0f * g * l2e;        // +2g*log2e

    // ---- A operands (X tile, 16 rows x 64 K), reused across 4 B tiles ----
    // A layout (16x32 f16): lanes 0-15 hold row l15 with K {0..7}U{16..23};
    // lanes 16-31 hold the same row with K {8..15}U{24..31}.
    const int oA = hi16 * 8;
    const _Float16* xh = Xh + (size_t)(mBase + l15) * DDIM;
    const _Float16* xl = Xl + (size_t)(mBase + l15) * DDIM;
    v16h ah0 = ld16(xh,  0 + oA,  0 + oA + 16);   // k-step 0 (K 0..31)
    v16h ah1 = ld16(xh, 32 + oA, 32 + oA + 16);   // k-step 1 (K 32..63)
    v16h al0 = ld16(xl,  0 + oA,  0 + oA + 16);
    v16h al1 = ld16(xl, 32 + oA, 32 + oA + 16);

    // per-lane folded row-norm constants: c[v] = -g*log2e * ||x_m||^2
    float c[8];
#pragma unroll
    for (int v = 0; v < 8; ++v) c[v] = gl * xsq[mBase + hi16 * 8 + v];

    // hoisted per-lane base pointers; j/v indexing folds into the signed
    // 24-bit instruction offset of global_load/global_store.
    const int n0 = nBase + l15;
    const int oB = hi16 * 8;    // B layout: lanes 0-15 K{0..7}U{16..23}, 16-31 K{8..15}U{24..31}
    const _Float16* yhp = Yh + (size_t)n0 * DDIM + oB;
    const _Float16* ylp = Yl + (size_t)n0 * DDIM + oB;
    const float*    ysp = ysq + n0;
    float*          op  = out + (size_t)(mBase + hi16 * 8) * NDIM + n0;

#pragma unroll
    for (int j = 0; j < 4; ++j) {
        const int jo = j * 16 * DDIM;           // 16 columns ahead in Y arrays
        v16h bh0 = ld16(yhp, jo +  0, jo + 16); // k-step 0
        v16h bh1 = ld16(yhp, jo + 32, jo + 48); // k-step 1
        v16h bl0 = ld16(ylp, jo +  0, jo + 16);
        v16h bl1 = ld16(ylp, jo + 32, jo + 48);

        v8f acc = {0.f, 0.f, 0.f, 0.f, 0.f, 0.f, 0.f, 0.f};
        // xh.yh (both k-steps)
        acc = __builtin_amdgcn_wmma_f32_16x16x32_f16(false, ah0, false, bh0,
                                                     (short)0, acc, false, false);
        acc = __builtin_amdgcn_wmma_f32_16x16x32_f16(false, ah1, false, bh1,
                                                     (short)0, acc, false, false);
        // xh.yl
        acc = __builtin_amdgcn_wmma_f32_16x16x32_f16(false, ah0, false, bl0,
                                                     (short)0, acc, false, false);
        acc = __builtin_amdgcn_wmma_f32_16x16x32_f16(false, ah1, false, bl1,
                                                     (short)0, acc, false, false);
        // xl.yh
        acc = __builtin_amdgcn_wmma_f32_16x16x32_f16(false, al0, false, bh0,
                                                     (short)0, acc, false, false);
        acc = __builtin_amdgcn_wmma_f32_16x16x32_f16(false, al1, false, bh1,
                                                     (short)0, acc, false, false);

        // ---- folded epilogue: exp2(min(k2*dot + (c[v]+u), 0)) ----
        const float u = gl * ysp[j * 16];
#pragma unroll
        for (int v = 0; v < 8; ++v) {
            float arg = fmaf(k2, acc[v], c[v] + u);
            arg = fminf(arg, 0.0f);             // == clamp sq_norm >= 0
            op[(size_t)v * NDIM + j * 16] = __builtin_amdgcn_exp2f(arg);
        }
    }
}

extern "C" void kernel_launch(void* const* d_in, const int* in_sizes, int n_in,
                              void* d_out, int out_size, void* d_ws, size_t ws_size,
                              hipStream_t stream) {
    const float* X     = (const float*)d_in[0];
    const float* Y     = (const float*)d_in[1];
    const float* gamma = (const float*)d_in[2];
    float* out = (float*)d_out;

    // workspace layout: [xsq 8192 f32][ysq 8192 f32][Xh][Xl][Yh][Yl] (f16 rows)
    float*    xsq = (float*)d_ws;
    float*    ysq = xsq + MDIM;
    _Float16* Xh  = (_Float16*)(ysq + NDIM);
    _Float16* Xl  = Xh + (size_t)MDIM * DDIM;
    _Float16* Yh  = Xl + (size_t)MDIM * DDIM;
    _Float16* Yl  = Yh + (size_t)NDIM * DDIM;

    // prep: one thread per row (X rows then Y rows)
    rbf_prep<<<(MDIM + NDIM) / 256, 256, 0, stream>>>(X, Y, xsq, ysq, Xh, Xl, Yh, Yl);

    // main: block tile 128(M) x 64(N), 8 waves of 16x64 each
    dim3 grid(NDIM / 64, MDIM / 128);
    rbf_main<<<grid, 256, 0, stream>>>(Xh, Xl, Yh, Yl, xsq, ysq, gamma, out);
}
